// MaskedAttention_9311489098275
// MI455X (gfx1250) — matrix operand order repaired and checked by reading, not compiled, not run on hardware
//
#include <hip/hip_runtime.h>
#include <hip/hip_bf16.h>

// ---------------------------------------------------------------------------
// MI455X (gfx1250) causal multi-head attention forward.
// Matmuls: v_wmma_f32_16x16x32_bf16 (fp32 accumulate).
// Tile staging: global_load_async_to_lds_b128 (ASYNCcnt), double-buffered GEMM.
// ---------------------------------------------------------------------------

typedef __attribute__((ext_vector_type(16))) __bf16 v16bf;
typedef __attribute__((ext_vector_type(8)))  float  v8f;

#define NH     16
#define TSEQ   2048
#define CDIM   1024
#define DHEAD  64
#define BATCH  2
#define MTOT   (BATCH * TSEQ)     // 4096 rows for the projection GEMMs

// ---------------------------------------------------------------------------
// WMMA + async helpers (CDNA5 layouts, wave32)
// ---------------------------------------------------------------------------
__device__ __forceinline__ v8f wmma_bf16(v16bf a, v16bf b, v8f c) {
  return __builtin_amdgcn_wmma_f32_16x16x32_bf16(false, a, false, b,
                                                 (short)0, c, false, false);
}

// Generic pointers to LDS carry the LDS byte address in bits [31:0].
__device__ __forceinline__ uint32_t lds_addr(const void* p) {
  return (uint32_t)(uintptr_t)p;
}

// 16B global -> LDS async copy, tracked by ASYNCcnt (no VGPR staging).
__device__ __forceinline__ void async_load_b128(uint32_t dst_lds,
                                                const void* src) {
  asm volatile("global_load_async_to_lds_b128 %0, %1, off"
               :: "v"(dst_lds), "v"((uint64_t)(uintptr_t)src)
               : "memory");
}

// A-operand fragment: 16(M) x 32(K) bf16.  Lane: M = lane&15, half = lane>>4.
__device__ __forceinline__ v16bf load_frag_row16(const __bf16* base, int ld,
                                                 int lm, int lh) {
  union { uint4 u[2]; v16bf v; } f;
  const __bf16* p = base + lm * ld + lh * 8;
  f.u[0] = *(const uint4*)(p);
  f.u[1] = *(const uint4*)(p + 16);
  return f.v;
}

// B-operand fragment from an [n][k] (n-major, k-contiguous) tile: 32(K) x 16(N).
__device__ __forceinline__ v16bf load_frag_col16(const __bf16* base, int ld,
                                                 int lm, int lh) {
  union { uint4 u[2]; v16bf v; } f;
  const __bf16* p = base + lm * ld + lh * 8;
  f.u[0] = *(const uint4*)(p);
  f.u[1] = *(const uint4*)(p + 16);
  return f.v;
}

__device__ __forceinline__ v8f v8f_zero() {
  v8f z = {0.f, 0.f, 0.f, 0.f, 0.f, 0.f, 0.f, 0.f};
  return z;
}

// ---------------------------------------------------------------------------
// fp32 -> bf16 bulk convert (n is a multiple of 4)
// ---------------------------------------------------------------------------
__global__ __launch_bounds__(256) void cvt_f32_to_bf16(
    const float* __restrict__ s, __bf16* __restrict__ d, int n) {
  int i = (blockIdx.x * 256 + threadIdx.x) * 4;
  if (i >= n) return;
  const float4 f = *(const float4*)(s + i);
  union { __bf16 b[4]; uint2 u; } c;
  c.b[0] = (__bf16)f.x; c.b[1] = (__bf16)f.y;
  c.b[2] = (__bf16)f.z; c.b[3] = (__bf16)f.w;
  *(uint2*)(d + i) = c.u;
}

// ---------------------------------------------------------------------------
// NT GEMM:  out[m,n] = sum_k A[m,k] * W[n,k] + bias[n]
// MODE 0: bf16 out -> [B][NH][T][D]   (Q, K)
// MODE 1: bf16 out -> [B][NH][D][T]   (V transposed)
// MODE 2: fp32 out -> plain [M][N]    (final projection)
// Block: 256 thr (8 waves, 2x4), macro tile 128x128, BK=32.
// LDS double-buffered; tiles staged with async-to-LDS, ASYNCcnt pipelined.
// ---------------------------------------------------------------------------
template <int MODE>
__global__ __launch_bounds__(256) void gemm_nt(
    const __bf16* __restrict__ A, const __bf16* __restrict__ W,
    const float* __restrict__ bias, void* __restrict__ outp) {
  constexpr int Kd  = CDIM;
  constexpr int LDT = 40;                       // 80B stride: 16B-aligned, padded
  __shared__ __align__(16) __bf16 sA[2][128 * LDT];
  __shared__ __align__(16) __bf16 sB[2][128 * LDT];

  const int tid  = threadIdx.x;
  const int bm   = blockIdx.x * 128, bn = blockIdx.y * 128;
  const int wave = tid >> 5, lane = tid & 31;
  const int wr   = wave >> 2, wc = wave & 3;    // wave -> (64-row, 32-col) region
  const int lm   = lane & 15, lh = lane >> 4;

  int rowL[2], ccL[2];
#pragma unroll
  for (int p = 0; p < 2; ++p) {
    int idx  = tid + p * 256;                   // 512 chunks of 16B cover 128x32
    rowL[p]  = idx >> 2;
    ccL[p]   = idx & 3;
  }

  // Issue one 128x32 A tile + 128x32 W tile into LDS buffer `buf` (async).
  auto issue_tile = [&](int k0, int buf) {
#pragma unroll
    for (int p = 0; p < 2; ++p) {
      async_load_b128(lds_addr(sA[buf] + rowL[p] * LDT + ccL[p] * 8),
                      A + (size_t)(bm + rowL[p]) * Kd + k0 + ccL[p] * 8);
      async_load_b128(lds_addr(sB[buf] + rowL[p] * LDT + ccL[p] * 8),
                      W + (size_t)(bn + rowL[p]) * Kd + k0 + ccL[p] * 8);
    }
  };

  issue_tile(0, 0);

  v8f acc[4][2];
#pragma unroll
  for (int i = 0; i < 4; ++i)
#pragma unroll
    for (int j = 0; j < 2; ++j) acc[i][j] = v8f_zero();

  for (int k0 = 0; k0 < Kd; k0 += 32) {
    const int buf = (k0 >> 5) & 1;
    if (k0 + 32 < Kd) {
      issue_tile(k0 + 32, buf ^ 1);             // prefetch next tile into other buf
      // Async loads complete in order: <=4 outstanding => current tile resident.
      asm volatile("s_wait_asynccnt 0x4" ::: "memory");
    } else {
      asm volatile("s_wait_asynccnt 0x0" ::: "memory");
    }
    __syncthreads();

    v16bf af[4], bfr[2];
#pragma unroll
    for (int i = 0; i < 4; ++i)
      af[i] = load_frag_row16(sA[buf] + (wr * 64 + i * 16) * LDT, LDT, lm, lh);
#pragma unroll
    for (int j = 0; j < 2; ++j)
      bfr[j] = load_frag_col16(sB[buf] + (wc * 32 + j * 16) * LDT, LDT, lm, lh);
#pragma unroll
    for (int i = 0; i < 4; ++i)
#pragma unroll
      for (int j = 0; j < 2; ++j) acc[i][j] = wmma_bf16(af[i], bfr[j], acc[i][j]);
    __syncthreads();                            // before buf is refilled (i+2)
  }

  // Epilogue: bias + scatter per C/D layout (elem e -> row e+8*half, col lane&15)
#pragma unroll
  for (int i = 0; i < 4; ++i) {
#pragma unroll
    for (int j = 0; j < 2; ++j) {
      const int n   = bn + wc * 32 + j * 16 + lm;
      const float bv = bias[n];
#pragma unroll
      for (int e = 0; e < 8; ++e) {
        const int m = bm + wr * 64 + i * 16 + e + lh * 8;
        float v = acc[i][j][e] + bv;
        if (MODE == 2) {
          ((float*)outp)[(size_t)m * CDIM + n] = v;
        } else {
          const int b = m >> 11, t = m & (TSEQ - 1);
          const int h = n >> 6, d = n & 63;
          size_t idx;
          if (MODE == 0) idx = ((size_t)(b * NH + h) * TSEQ + t) * DHEAD + d;
          else           idx = ((size_t)(b * NH + h) * DHEAD + d) * TSEQ + t;
          ((__bf16*)outp)[idx] = (__bf16)v;
        }
      }
    }
  }
}

// ---------------------------------------------------------------------------
// Flash-style causal attention.
//   Q,K: [B][NH][T][D] bf16.  Vt: [B][NH][D][T] bf16.  attn: [B][T][C] bf16.
// Block = (b, h, 64 q-rows); 4 waves x 16 rows; 32 keys / step.
// K/V tiles staged with async-to-LDS.
// ---------------------------------------------------------------------------
__global__ __launch_bounds__(128) void attn_fwd(
    const __bf16* __restrict__ Q, const __bf16* __restrict__ K,
    const __bf16* __restrict__ Vt, __bf16* __restrict__ attn) {
  constexpr int LDK = 80, LDV = 40, LDP = 40;   // all strides 16B-aligned
  __shared__ __align__(16) __bf16 sK[32 * LDK];       // 32 keys x 64 d
  __shared__ __align__(16) __bf16 sV[64 * LDV];       // 64 d x 32 keys (V^T)
  __shared__ __align__(16) __bf16 sP[4 * 16 * LDP];   // per-wave P scratch

  const int tid = threadIdx.x, wave = tid >> 5, lane = tid & 31;
  const int lm = lane & 15, lh = lane >> 4;
  const int b = blockIdx.z, h = blockIdx.y, q0 = blockIdx.x * 64;
  const int qr = q0 + wave * 16;

  const __bf16* Qbase = Q + ((size_t)(b * NH + h) * TSEQ + qr) * DHEAD;
  const __bf16* Kbase = K + (size_t)(b * NH + h) * TSEQ * DHEAD;
  const __bf16* Vbase = Vt + (size_t)(b * NH + h) * DHEAD * TSEQ;
  __bf16* sPw = sP + wave * 16 * LDP;

  const v16bf qf0 = load_frag_row16(Qbase, DHEAD, lm, lh);       // d 0..31
  const v16bf qf1 = load_frag_row16(Qbase + 32, DHEAD, lm, lh);  // d 32..63

  float mrow[8], lrow[8];
  v8f o[4];
#pragma unroll
  for (int e = 0; e < 8; ++e) { mrow[e] = -3.0e38f; lrow[e] = 0.f; }
#pragma unroll
  for (int dt = 0; dt < 4; ++dt) o[dt] = v8f_zero();

  for (int s0 = 0; s0 < q0 + 64; s0 += 32) {
    // Async stage: K 32x64 and V^T 64x32 (16B chunks, coalesced).
#pragma unroll
    for (int p = 0; p < 2; ++p) {
      const int idx = tid + p * 128;
      const int r = idx >> 3, cc = idx & 7;     // K: 8 chunks per 64-elem row
      async_load_b128(lds_addr(sK + r * LDK + cc * 8),
                      Kbase + (size_t)(s0 + r) * DHEAD + cc * 8);
      const int rv = idx >> 2, cv = idx & 3;    // V^T: 4 chunks per 32-elem row
      async_load_b128(lds_addr(sV + rv * LDV + cv * 8),
                      Vbase + (size_t)rv * TSEQ + s0 + cv * 8);
    }
    asm volatile("s_wait_asynccnt 0x0" ::: "memory");
    __syncthreads();

    if (s0 <= qr + 15) {                        // wave has unmasked work here
      v8f sacc[2];
#pragma unroll
      for (int sc = 0; sc < 2; ++sc) {
        sacc[sc] = v8f_zero();
        sacc[sc] = wmma_bf16(qf0,
            load_frag_col16(sK + sc * 16 * LDK, LDK, lm, lh), sacc[sc]);
        sacc[sc] = wmma_bf16(qf1,
            load_frag_col16(sK + sc * 16 * LDK + 32, LDK, lm, lh), sacc[sc]);
      }

      float fac[8];
#pragma unroll
      for (int e = 0; e < 8; ++e) {
        float v0 = sacc[0][e] * 0.125f;         // 1/sqrt(64)
        float v1 = sacc[1][e] * 0.125f;
        const int qg = qr + e + lh * 8;
        if (s0 + lm > qg)      v0 = -3.0e38f;   // causal mask
        if (s0 + 16 + lm > qg) v1 = -3.0e38f;
        float rm = fmaxf(v0, v1);
#pragma unroll
        for (int x = 1; x < 16; x <<= 1) rm = fmaxf(rm, __shfl_xor(rm, x, 32));
        const float mnew = fmaxf(mrow[e], rm);
        fac[e] = __expf(mrow[e] - mnew);
        const float p0 = __expf(v0 - mnew);
        const float p1 = __expf(v1 - mnew);
        sPw[(e + lh * 8) * LDP + lm]      = (__bf16)p0;
        sPw[(e + lh * 8) * LDP + 16 + lm] = (__bf16)p1;
        float ps = p0 + p1;
#pragma unroll
        for (int x = 1; x < 16; x <<= 1) ps += __shfl_xor(ps, x, 32);
        lrow[e] = lrow[e] * fac[e] + ps;
        mrow[e] = mnew;
      }
      asm volatile("s_wait_dscnt 0x0" ::: "memory");  // P visible before re-read

      const v16bf pf = load_frag_row16(sPw, LDP, lm, lh);
#pragma unroll
      for (int dt = 0; dt < 4; ++dt) {
        const v16bf vf = load_frag_col16(sV + dt * 16 * LDV, LDV, lm, lh);
#pragma unroll
        for (int e = 0; e < 8; ++e) o[dt][e] *= fac[e];
        o[dt] = wmma_bf16(pf, vf, o[dt]);
      }
    }
    __syncthreads();                            // LDS reused next step
  }

#pragma unroll
  for (int e = 0; e < 8; ++e) {
    const float inv = 1.0f / lrow[e];
    const int t = qr + e + lh * 8;
#pragma unroll
    for (int dt = 0; dt < 4; ++dt) {
      const int c = h * DHEAD + dt * 16 + lm;
      attn[(size_t)(b * TSEQ + t) * CDIM + c] = (__bf16)(o[dt][e] * inv);
    }
  }
}

// ---------------------------------------------------------------------------
// Launch
// ---------------------------------------------------------------------------
extern "C" void kernel_launch(void* const* d_in, const int* in_sizes, int n_in,
                              void* d_out, int out_size, void* d_ws,
                              size_t ws_size, hipStream_t stream) {
  const float* x  = (const float*)d_in[0];
  const float* Wq = (const float*)d_in[1];
  const float* bq = (const float*)d_in[2];
  const float* Wk = (const float*)d_in[3];
  const float* bk = (const float*)d_in[4];
  const float* Wv = (const float*)d_in[5];
  const float* bv = (const float*)d_in[6];
  const float* Wp = (const float*)d_in[7];
  const float* bp = (const float*)d_in[8];

  constexpr size_t XB = (size_t)MTOT * CDIM;          // 4,194,304
  constexpr size_t WB = (size_t)CDIM * CDIM;          // 1,048,576
  char* ws = (char*)d_ws;
  __bf16* xb    = (__bf16*)(ws);                 ws += XB * 2;
  __bf16* wqb   = (__bf16*)(ws);                 ws += WB * 2;
  __bf16* wkb   = (__bf16*)(ws);                 ws += WB * 2;
  __bf16* wvb   = (__bf16*)(ws);                 ws += WB * 2;
  __bf16* wpb   = (__bf16*)(ws);                 ws += WB * 2;
  __bf16* qb    = (__bf16*)(ws);                 ws += XB * 2;
  __bf16* kb    = (__bf16*)(ws);                 ws += XB * 2;
  __bf16* vtb   = (__bf16*)(ws);                 ws += XB * 2;
  __bf16* attnb = (__bf16*)(ws);                 ws += XB * 2;   // ~48 MB total

  cvt_f32_to_bf16<<<(int)(XB / 4 / 256), 256, 0, stream>>>(x, xb, (int)XB);
  cvt_f32_to_bf16<<<(int)(WB / 4 / 256), 256, 0, stream>>>(Wq, wqb, (int)WB);
  cvt_f32_to_bf16<<<(int)(WB / 4 / 256), 256, 0, stream>>>(Wk, wkb, (int)WB);
  cvt_f32_to_bf16<<<(int)(WB / 4 / 256), 256, 0, stream>>>(Wv, wvb, (int)WB);
  cvt_f32_to_bf16<<<(int)(WB / 4 / 256), 256, 0, stream>>>(Wp, wpb, (int)WB);

  dim3 gg(MTOT / 128, CDIM / 128);
  gemm_nt<0><<<gg, 256, 0, stream>>>(xb, wqb, bq, (void*)qb);
  gemm_nt<0><<<gg, 256, 0, stream>>>(xb, wkb, bk, (void*)kb);
  gemm_nt<1><<<gg, 256, 0, stream>>>(xb, wvb, bv, (void*)vtb);

  dim3 ga(TSEQ / 64, NH, BATCH);
  attn_fwd<<<ga, 128, 0, stream>>>(qb, kb, vtb, attnb);

  gemm_nt<2><<<gg, 256, 0, stream>>>(attnb, wpb, bp, d_out);
}
